// VVAELatentActionVQVAE_33646773797650
// MI455X (gfx1250) — matrix operand order repaired and checked by reading, not compiled, not run on hardware
//
#include <hip/hip_runtime.h>

// =====================================================================
// VVAE latent-action VQVAE forward for MI455X (gfx1250, wave32, WMMA).
// All GEMMs run through v_wmma_f32_16x16x32_bf16 (bf16 in, f32 acc).
// =====================================================================

typedef unsigned short u16;
typedef __bf16 bf16_t;
typedef bf16_t v16bf __attribute__((ext_vector_type(16)));
typedef u16   v16u  __attribute__((ext_vector_type(16)));
typedef float v8f   __attribute__((ext_vector_type(8)));

#define TILE_M 64
#define TILE_N 32
#define TILE_K 32
#define GEMM_THREADS 128
#define ACT_NONE 0
#define ACT_GELU 1

__device__ __forceinline__ u16 f32_to_bf16(float f) {
  unsigned int x = __float_as_uint(f);
  x += 0x7FFFu + ((x >> 16) & 1u);      // round-to-nearest-even
  return (u16)(x >> 16);
}
__device__ __forceinline__ float gelu_f(float v) {
  float t = 0.7978845608028654f * (v + 0.044715f * v * v * v);
  return 0.5f * v * (1.0f + tanhf(t));
}

// ---------------------------------------------------------------------
// Generic bf16 GEMM: C = act(scale * A(MxK) * B(KxN) + bias + res)
// BT=1 => B is stored [N,K] row-major (i.e. compute A * B^T).
// Block tile 64x32: 4 waves x (16 rows), 2 WMMAs per wave per K-step.
// Interior tiles: int4 (8xbf16) vector loads, b128 LDS stores.
// Edge tiles: clamped unconditional loads + select (no exec branches).
// Two-level (batch, head) z-strides allow batched attention GEMMs.
// Fragment packing: contiguous k = 16*(lane/16) + e, same order for A and B.
// ---------------------------------------------------------------------
template <int BT>
__global__ __launch_bounds__(GEMM_THREADS)
void gemm_bf16_wmma(const u16* __restrict__ Ag, const u16* __restrict__ Bg,
                    const float* __restrict__ bias, const float* __restrict__ res,
                    float* __restrict__ C, u16* __restrict__ Cb,
                    int M, int N, int K, int lda, int ldb, int ldc,
                    float scale, int act, int Hd,
                    long long sAb, long long sAh, long long sBb, long long sBh,
                    long long sCb, long long sCh)
{
  __shared__ alignas(16) u16 As[TILE_M][TILE_K + 8];
  __shared__ alignas(16) u16 Bs[TILE_N][TILE_K + 8];  // BT=1: [n][k]; BT=0: [k][n]

  const int z  = blockIdx.z;
  const int bz = z / Hd, hz = z % Hd;
  const u16* A = Ag + bz * sAb + hz * sAh;
  const u16* B = Bg + bz * sBb + hz * sBh;
  const long long coff = bz * sCb + hz * sCh;

  const int tid  = threadIdx.x;
  const int wave = tid >> 5;
  const int lane = tid & 31;
  const int l15  = lane & 15;
  const int khalf = lane >> 4;
  const int m0 = blockIdx.y * TILE_M;
  const int n0 = blockIdx.x * TILE_N;

  const bool alA = ((lda & 7) == 0);
  const bool alB = ((ldb & 7) == 0);

  v8f acc0 = {}, acc1 = {};

  for (int k0 = 0; k0 < K; k0 += TILE_K) {
    const bool fullK = (k0 + TILE_K <= K);

    // ---------------- A tile (64 x 32) ----------------
    if (fullK && alA && (m0 + TILE_M <= M)) {
#pragma unroll
      for (int i = 0; i < 2; ++i) {
        int c = tid + i * GEMM_THREADS;     // 256 chunks of 8 bf16
        int r = c >> 2, c8 = (c & 3) << 3;
        const int4 v = *(const int4*)(A + (long long)(m0 + r) * lda + (k0 + c8));
        *(int4*)(&As[r][c8]) = v;
      }
    } else {
#pragma unroll
      for (int i = 0; i < 16; ++i) {
        int e = tid + GEMM_THREADS * i;
        int r = e >> 5, cc = e & 31;
        int gr = m0 + r, gc = k0 + cc;
        int rr = gr < M ? gr : M - 1;
        int rc = gc < K ? gc : K - 1;
        u16 v = A[(long long)rr * lda + rc];
        if (gr >= M || gc >= K) v = 0;
        As[r][cc] = v;
      }
    }

    // ---------------- B tile (32 x 32) ----------------
    if (BT) {
      if (fullK && alB && (n0 + TILE_N <= N)) {
        int r = tid >> 2, c8 = (tid & 3) << 3;
        const int4 v = *(const int4*)(B + (long long)(n0 + r) * ldb + (k0 + c8));
        *(int4*)(&Bs[r][c8]) = v;
      } else {
#pragma unroll
        for (int i = 0; i < 8; ++i) {
          int e = tid + GEMM_THREADS * i;
          int n = e >> 5, cc = e & 31;
          int gn = n0 + n, gc = k0 + cc;
          int rn = gn < N ? gn : N - 1;
          int rc = gc < K ? gc : K - 1;
          u16 v = B[(long long)rn * ldb + rc];
          if (gn >= N || gc >= K) v = 0;
          Bs[n][cc] = v;
        }
      }
    } else {
      if (fullK && alB && (n0 + TILE_N <= N)) {
        int r = tid >> 2, c8 = (tid & 3) << 3;   // r: k-row, c8: n-offset
        const int4 v = *(const int4*)(B + (long long)(k0 + r) * ldb + (n0 + c8));
        *(int4*)(&Bs[r][c8]) = v;
      } else {
#pragma unroll
        for (int i = 0; i < 8; ++i) {
          int e = tid + GEMM_THREADS * i;
          int kk = e >> 5, nn = e & 31;
          int gk = k0 + kk, gn = n0 + nn;
          int rk = gk < K ? gk : K - 1;
          int rn = gn < N ? gn : N - 1;
          u16 v = B[(long long)rk * ldb + rn];
          if (gk >= K || gn >= N) v = 0;
          Bs[kk][nn] = v;
        }
      }
    }

    // prefetch next A K-tile (lowers to global_prefetch_b8 on gfx1250)
    if (k0 + TILE_K < K) {
      int gr = m0 + (tid >> 1);
      if (gr < M) __builtin_prefetch(A + (long long)gr * lda + (k0 + TILE_K), 0, 1);
    }
    __syncthreads();

    // Fragment packing: k = khalf*16 + e (contiguous), same order for A and B
    union { v16u u; v16bf h; } fa, fb0, fb1;
    const int mrow = wave * 16 + l15;
#pragma unroll
    for (int e = 0; e < 16; ++e) {
      int vv = e >> 1, s = e & 1;
      int kk = khalf * 16 + e + 0 * (vv + s);
      fa.u[e] = As[mrow][kk];
      if (BT) {
        fb0.u[e] = Bs[l15][khalf * 16 + e];
        fb1.u[e] = Bs[16 + l15][khalf * 16 + e];
      } else {
        fb0.u[e] = Bs[khalf * 16 + e][l15];
        fb1.u[e] = Bs[khalf * 16 + e][16 + l15];
      }
    }
    acc0 = __builtin_amdgcn_wmma_f32_16x16x32_bf16(false, fa.h, false, fb0.h,
                                                   (short)0, acc0, false, false);
    acc1 = __builtin_amdgcn_wmma_f32_16x16x32_bf16(false, fa.h, false, fb1.h,
                                                   (short)0, acc1, false, false);
    __syncthreads();
  }

  // D layout: VGPR r -> row m0 + wave*16 + r + (lane>=16 ? 8 : 0), col = lane&15
#pragma unroll
  for (int j = 0; j < 2; ++j) {
    const v8f a = j ? acc1 : acc0;
    const int col = n0 + j * 16 + l15;
    if (col < N) {
      const float bv = bias ? bias[col] : 0.0f;
#pragma unroll
      for (int r = 0; r < 8; ++r) {
        int row = m0 + wave * 16 + r + (khalf << 3);
        if (row < M) {
          long long oi = coff + (long long)row * ldc + col;
          float v = a[r] * scale + bv;
          if (res) v += res[oi];
          if (act == ACT_GELU) v = gelu_f(v);
          if (C)  C[oi]  = v;
          if (Cb) Cb[oi] = f32_to_bf16(v);
        }
      }
    }
  }
}

// ---------------------------------------------------------------------
// fp32 -> bf16 bulk convert
// ---------------------------------------------------------------------
__global__ __launch_bounds__(256)
void f32_to_bf16_kernel(const float* __restrict__ src, u16* __restrict__ dst,
                        long long n) {
  long long i = (long long)blockIdx.x * 256 + threadIdx.x;
  long long stride = (long long)gridDim.x * 256;
  for (; i < n; i += stride) dst[i] = f32_to_bf16(src[i]);
}

// ---------------------------------------------------------------------
// LayerNorm over last dim (cols). g/b nullable. Dual bf16/f32 outputs
// (either nullable). Safe in-place when Yf == X (one block per row).
// ---------------------------------------------------------------------
__global__ __launch_bounds__(256)
void layernorm_kernel(const float* __restrict__ X, const float* __restrict__ g,
                      const float* __restrict__ b, u16* __restrict__ Yb,
                      float* __restrict__ Yf, int cols) {
  const int r = blockIdx.x;
  const float* xr = X + (long long)r * cols;
  __shared__ float red[256];
  float s = 0.f, s2 = 0.f;
  for (int c = threadIdx.x; c < cols; c += 256) { float v = xr[c]; s += v; s2 += v * v; }
  red[threadIdx.x] = s; __syncthreads();
  for (int o = 128; o > 0; o >>= 1) { if (threadIdx.x < o) red[threadIdx.x] += red[threadIdx.x + o]; __syncthreads(); }
  float mean = red[0] / cols; __syncthreads();
  red[threadIdx.x] = s2; __syncthreads();
  for (int o = 128; o > 0; o >>= 1) { if (threadIdx.x < o) red[threadIdx.x] += red[threadIdx.x + o]; __syncthreads(); }
  float var = red[0] / cols - mean * mean;
  float inv = rsqrtf(var + 1e-5f);
  for (int c = threadIdx.x; c < cols; c += 256) {
    float v = (xr[c] - mean) * inv;
    if (g) v = v * g[c] + b[c];
    long long oi = (long long)r * cols + c;
    if (Yf) Yf[oi] = v;
    if (Yb) Yb[oi] = f32_to_bf16(v);
  }
}

// ---------------------------------------------------------------------
// Row softmax with optional frame-causal mask (mask if k_frame >= q_frame+2)
// grid = (Lq, Z). Row pitch Lkp (padded); emits bf16 probabilities.
// ---------------------------------------------------------------------
__global__ __launch_bounds__(256)
void softmax_kernel(const float* __restrict__ S, u16* __restrict__ P,
                    int Lk, int Lkp, long long sS, int mask_n) {
  const int q = blockIdx.x, z = blockIdx.y;
  const float* row = S + (long long)z * sS + (long long)q * Lkp;
  u16* prow = P + (long long)z * sS + (long long)q * Lkp;
  __shared__ float red[256];
  const int qf = mask_n ? (q / mask_n) : 0;
  float mx = -3.0e38f;
  for (int k = threadIdx.x; k < Lk; k += 256) {
    float v = row[k];
    if (mask_n && (k / mask_n) >= qf + 2) v -= 1.0e9f;
    mx = fmaxf(mx, v);
  }
  red[threadIdx.x] = mx; __syncthreads();
  for (int o = 128; o > 0; o >>= 1) { if (threadIdx.x < o) red[threadIdx.x] = fmaxf(red[threadIdx.x], red[threadIdx.x + o]); __syncthreads(); }
  mx = red[0]; __syncthreads();
  float sum = 0.f;
  for (int k = threadIdx.x; k < Lk; k += 256) {
    float v = row[k];
    if (mask_n && (k / mask_n) >= qf + 2) v -= 1.0e9f;
    sum += expf(v - mx);
  }
  red[threadIdx.x] = sum; __syncthreads();
  for (int o = 128; o > 0; o >>= 1) { if (threadIdx.x < o) red[threadIdx.x] += red[threadIdx.x + o]; __syncthreads(); }
  float inv = 1.0f / red[0];
  for (int k = threadIdx.x; k < Lk; k += 256) {
    float v = row[k];
    if (mask_n && (k / mask_n) >= qf + 2) v -= 1.0e9f;
    prow[k] = f32_to_bf16(expf(v - mx) * inv);
  }
  for (int k = Lk + threadIdx.x; k < Lkp; k += 256) prow[k] = 0;  // zero pad
}

// x[(b,t,n),c] += spos[n,c] + tpos[t,c]   (encoder, rows = (b*T+t)*N+n)
__global__ __launch_bounds__(256)
void addpos_enc_kernel(float* __restrict__ x, const float* __restrict__ spos,
                       const float* __restrict__ tpos, long long total) {
  long long i = (long long)blockIdx.x * 256 + threadIdx.x;
  long long stride = (long long)gridDim.x * 256;
  for (; i < total; i += stride) {
    long long row = i >> 9; int c = (int)(i & 511);
    int t = (int)((row >> 8) & 7); int n = (int)(row & 255);
    x[i] += spos[n * 512 + c] + tpos[t * 512 + c];
  }
}

// frames[b*T+t, c] = mean_n x[(b*T+t)*N+n, c]
__global__ __launch_bounds__(256)
void meanpool_kernel(const float* __restrict__ X, float* __restrict__ F,
                     u16* __restrict__ Fb) {
  const int f = blockIdx.x;
  for (int c = threadIdx.x; c < 512; c += 256) {
    float s = 0.f;
    for (int n = 0; n < 256; ++n) s += X[((long long)(f * 256 + n)) * 512 + c];
    float v = s * (1.0f / 256.0f);
    F[f * 512 + c] = v;
    Fb[f * 512 + c] = f32_to_bf16(v);
  }
}

// tr[b*7 + (t-1)] = frames_p[b*8 + t], t = 1..7  (bf16 for the mu GEMM)
__global__ __launch_bounds__(256)
void gather_tr_kernel(const float* __restrict__ Fp, u16* __restrict__ Tr) {
  int i = blockIdx.x; int b = i / 7, t = i % 7 + 1;
  for (int c = threadIdx.x; c < 512; c += 256)
    Tr[(long long)i * 512 + c] = f32_to_bf16(Fp[(long long)(b * 8 + t) * 512 + c]);
}

// VQ: for each 128-d vector, argmin_j ||cb_j||^2 - 2 c.cb_j ; emit cb[best] bf16
__global__ __launch_bounds__(256)
void quantize_kernel(const float* __restrict__ mu, const float* __restrict__ cb,
                     u16* __restrict__ qb, int Kcb, int Dm) {
  const int i = blockIdx.x;
  __shared__ float sc[128];
  for (int d = threadIdx.x; d < Dm; d += 256) sc[d] = mu[(long long)i * Dm + d];
  __syncthreads();
  float best = 3.4e38f; int bi = 0;
  for (int j = threadIdx.x; j < Kcb; j += 256) {
    const float* cr = cb + (long long)j * Dm;
    float dot = 0.f, nrm = 0.f;
    for (int d = 0; d < Dm; ++d) { float w = cr[d]; dot += sc[d] * w; nrm += w * w; }
    float dist = nrm - 2.0f * dot;
    if (dist < best) { best = dist; bi = j; }
  }
  __shared__ float rb[256]; __shared__ int ri[256];
  rb[threadIdx.x] = best; ri[threadIdx.x] = bi; __syncthreads();
  for (int o = 128; o > 0; o >>= 1) {
    if (threadIdx.x < o && rb[threadIdx.x + o] < rb[threadIdx.x]) {
      rb[threadIdx.x] = rb[threadIdx.x + o]; ri[threadIdx.x] = ri[threadIdx.x + o];
    }
    __syncthreads();
  }
  int idx = ri[0];
  for (int d = threadIdx.x; d < Dm; d += 256)
    qb[(long long)i * Dm + d] = f32_to_bf16(cb[(long long)idx * Dm + d]);
}

// mem[b, t*259 + n] = bf16( zp[(b*7+t)*256+n] + spos[n] + tpos[t] )
__global__ __launch_bounds__(256)
void scatter_mem_z_kernel(const float* __restrict__ zp, const float* __restrict__ spos,
                          const float* __restrict__ tpos, u16* __restrict__ mem,
                          long long total) {
  long long i = (long long)blockIdx.x * 256 + threadIdx.x;
  long long stride = (long long)gridDim.x * 256;
  for (; i < total; i += stride) {
    long long row = i >> 9; int c = (int)(i & 511);
    int b = (int)(row / 1792); int k = (int)(row % 1792);
    int t = k / 256, n = k % 256;
    float v = zp[i] + spos[n * 512 + c] + tpos[t * 512 + c];
    long long orow = (long long)b * 1813 + t * 259 + n;
    mem[orow * 512 + c] = f32_to_bf16(v);
  }
}

// mem[b, t*259 + 256 + j] = bf16( cln[(b*21 + t*3 + j)] + tpos[t] )
__global__ __launch_bounds__(256)
void scatter_mem_c_kernel(const float* __restrict__ cln, const float* __restrict__ tpos,
                          u16* __restrict__ mem, long long total) {
  long long i = (long long)blockIdx.x * 256 + threadIdx.x;
  long long stride = (long long)gridDim.x * 256;
  for (; i < total; i += stride) {
    long long row = i >> 9; int c = (int)(i & 511);
    int b = (int)(row / 21); int k = (int)(row % 21);
    int t = k / 3, j = k % 3;
    float v = cln[i] + tpos[t * 512 + c];
    long long orow = (long long)b * 1813 + t * 259 + 256 + j;
    mem[orow * 512 + c] = f32_to_bf16(v);
  }
}

// q[b*256+n, c] = query_embed[n, c] + qpos[n, c]
__global__ __launch_bounds__(256)
void init_query_kernel(float* __restrict__ q, const float* __restrict__ qe,
                       const float* __restrict__ qpos, long long total) {
  long long i = (long long)blockIdx.x * 256 + threadIdx.x;
  long long stride = (long long)gridDim.x * 256;
  for (; i < total; i += stride) {
    long long row = i >> 9; int c = (int)(i & 511);
    int n = (int)(row & 255);
    q[i] = qe[n * 512 + c] + qpos[n * 512 + c];
  }
}

// =====================================================================
// Host orchestration
// =====================================================================
static inline int cdiv_i(int a, int b) { return (a + b - 1) / b; }

static void gemm(hipStream_t st, bool bt,
                 const u16* A, const u16* B, const float* bias, const float* res,
                 float* C, u16* Cb, int M, int N, int K,
                 int lda, int ldb, int ldc, float scale, int act,
                 int Z = 1, int Hd = 1,
                 long long sAb = 0, long long sAh = 0,
                 long long sBb = 0, long long sBh = 0,
                 long long sCb = 0, long long sCh = 0) {
  dim3 g(cdiv_i(N, TILE_N), cdiv_i(M, TILE_M), Z);
  dim3 blk(GEMM_THREADS);
  if (bt)
    gemm_bf16_wmma<1><<<g, blk, 0, st>>>(A, B, bias, res, C, Cb, M, N, K, lda, ldb, ldc,
                                         scale, act, Hd, sAb, sAh, sBb, sBh, sCb, sCh);
  else
    gemm_bf16_wmma<0><<<g, blk, 0, st>>>(A, B, bias, res, C, Cb, M, N, K, lda, ldb, ldc,
                                         scale, act, Hd, sAb, sAh, sBb, sBh, sCb, sCh);
}

// d_in index map: assumes jax pytree flatten (dict keys sorted) of
// {'z_sequence', 'params':{'dec','enc'}, 'codebook'} -> 152 leaves.
enum {
  IN_Z = 0,
  DEC_BLK0 = 1,
  DB_LN1B = 0, DB_LN1G = 1, DB_LN2B = 2, DB_LN2G = 3, DB_LN3B = 4, DB_LN3G = 5,
  DB_M_B1 = 6, DB_M_B2 = 7, DB_M_W1 = 8, DB_M_W2 = 9,
  DB_S_BK = 10, DB_S_BO = 11, DB_S_BQ = 12, DB_S_BV = 13,
  DB_S_WK = 14, DB_S_WO = 15, DB_S_WQ = 16, DB_S_WV = 17,
  DB_X_BK = 18, DB_X_BO = 19, DB_X_BQ = 20, DB_X_BV = 21,
  DB_X_WK = 22, DB_X_WO = 23, DB_X_WQ = 24, DB_X_WV = 25,
  DEC_CODE_B = 79, DEC_CODE_W = 80, DEC_NORM_B = 81, DEC_NORM_G = 82,
  DEC_OUT_B = 83, DEC_OUT_W = 84, DEC_PATCH_B = 85, DEC_PATCH_W = 86,
  DEC_QPOS = 87, DEC_QEMB = 88, DEC_SPOS = 89, DEC_TPOS = 90,
  ENC_BLK0 = 91,
  EB_BK = 0, EB_BO = 1, EB_BQ = 2, EB_BV = 3, EB_WK = 4, EB_WO = 5, EB_WQ = 6, EB_WV = 7,
  EB_LN1B = 8, EB_LN1G = 9, EB_LN2B = 10, EB_LN2G = 11,
  EB_M_B1 = 12, EB_M_B2 = 13, EB_M_W1 = 14, EB_M_W2 = 15,
  ENC_LV_B = 139, ENC_LV_W = 140, ENC_MU_B = 141, ENC_MU_W = 142,
  ENC_NORM_B = 143, ENC_NORM_G = 144, ENC_PATCH_B = 145, ENC_PATCH_W = 146,
  ENC_POOL_B = 147, ENC_POOL_W = 148, ENC_SPOS = 149, ENC_TPOS = 150,
  IN_CODEBOOK = 151
};

extern "C" void kernel_launch(void* const* d_in, const int* in_sizes, int n_in,
                              void* d_out, int out_size, void* d_ws, size_t ws_size,
                              hipStream_t stream) {
  (void)in_sizes; (void)n_in; (void)out_size; (void)ws_size;
  const int E = 512, DH = 64, Hh = 8, MLPH = 2048, CDIM = 128, KCB = 512;
  const int Senc = 4096;          // B*T*N tokens
  const int Lenc = 2048;          // per-batch encoder sequence
  const int Mdec = 512;           // B*N decoder queries
  const int Lmem = 1813;          // Tp*(N+3)
  const int LMP  = 1824;          // Lmem padded to a multiple of 32
  const int Smem = 3626;          // B*Lmem

  auto F = [&](int i) -> const float* { return (const float*)d_in[i]; };

  size_t off = 0;
  auto alloc = [&](size_t bytes) -> void* {
    void* p = (char*)d_ws + off;
    off += (bytes + 255) & ~(size_t)255;
    return p;
  };
  auto toBf16 = [&](const float* src, long long n) -> u16* {
    u16* d = (u16*)alloc((size_t)n * sizeof(u16));
    int blocks = (int)((n + 255) / 256); if (blocks > 8192) blocks = 8192;
    f32_to_bf16_kernel<<<dim3(blocks), dim3(256), 0, stream>>>(src, d, n);
    return d;
  };
  auto ln = [&](const float* X, const float* g, const float* b,
                u16* Yb, float* Yf, int rows) {
    layernorm_kernel<<<dim3(rows), dim3(256), 0, stream>>>(X, g, b, Yb, Yf, E);
  };

  // ---------------- bf16 weight conversion ----------------
  u16* zb = toBf16(F(IN_Z), (long long)Senc * 1024);
  u16* encPW = toBf16(F(ENC_PATCH_W), 1024LL * E);
  u16 *eWq[3], *eWk[3], *eWv[3], *eWo[3], *eW1[3], *eW2[3];
  for (int k = 0; k < 3; ++k) {
    int base = ENC_BLK0 + 16 * k;
    eWq[k] = toBf16(F(base + EB_WQ), (long long)E * E);
    eWk[k] = toBf16(F(base + EB_WK), (long long)E * E);
    eWv[k] = toBf16(F(base + EB_WV), (long long)E * E);
    eWo[k] = toBf16(F(base + EB_WO), (long long)E * E);
    eW1[k] = toBf16(F(base + EB_M_W1), (long long)E * MLPH);
    eW2[k] = toBf16(F(base + EB_M_W2), (long long)MLPH * E);
  }
  u16* poolW = toBf16(F(ENC_POOL_W), (long long)E * E);
  u16* muW   = toBf16(F(ENC_MU_W), (long long)E * 384);
  u16* decPW = toBf16(F(DEC_PATCH_W), 1024LL * E);
  u16* codeW = toBf16(F(DEC_CODE_W), (long long)CDIM * E);
  u16* outW  = toBf16(F(DEC_OUT_W), (long long)E * 1024);
  u16 *dsWq[3], *dsWk[3], *dsWv[3], *dsWo[3];
  u16 *dxWq[3], *dxWk[3], *dxWv[3], *dxWo[3];
  u16 *dW1[3], *dW2[3];
  for (int k = 0; k < 3; ++k) {
    int base = DEC_BLK0 + 26 * k;
    dsWq[k] = toBf16(F(base + DB_S_WQ), (long long)E * E);
    dsWk[k] = toBf16(F(base + DB_S_WK), (long long)E * E);
    dsWv[k] = toBf16(F(base + DB_S_WV), (long long)E * E);
    dsWo[k] = toBf16(F(base + DB_S_WO), (long long)E * E);
    dxWq[k] = toBf16(F(base + DB_X_WQ), (long long)E * E);
    dxWk[k] = toBf16(F(base + DB_X_WK), (long long)E * E);
    dxWv[k] = toBf16(F(base + DB_X_WV), (long long)E * E);
    dxWo[k] = toBf16(F(base + DB_X_WO), (long long)E * E);
    dW1[k]  = toBf16(F(base + DB_M_W1), (long long)E * MLPH);
    dW2[k]  = toBf16(F(base + DB_M_W2), (long long)MLPH * E);
  }

  // ---------------- activation buffers ----------------
  float* x  = (float*)alloc((size_t)Senc * E * 4);          // encoder residual stream
  u16* hb   = (u16*)alloc((size_t)Senc * E * 2);            // LN output (bf16)
  u16* qb   = (u16*)alloc((size_t)Senc * E * 2);
  u16* kb   = (u16*)alloc((size_t)Senc * E * 2);
  u16* vb   = (u16*)alloc((size_t)Senc * E * 2);
  u16* ob   = (u16*)alloc((size_t)Senc * E * 2);            // attention context
  u16* mlpb = (u16*)alloc((size_t)Senc * MLPH * 2);         // GELU hidden
  float* Sb = (float*)alloc((size_t)16 * 256 * LMP * 4);    // >= 2048*2048*4 too
  u16* Pb   = (u16*)alloc((size_t)16 * 256 * LMP * 2);
  float* frames  = (float*)alloc(16 * E * 4);
  u16* framesB   = (u16*)alloc(16 * E * 2);
  float* framesP = (float*)alloc(16 * E * 4);
  u16* trB       = (u16*)alloc(14 * E * 2);
  float* mu      = (float*)alloc(14 * 384 * 4);
  u16* qcB       = (u16*)alloc(42 * CDIM * 2);
  float* zp      = (float*)alloc((size_t)3584 * E * 4);
  float* cfull   = (float*)alloc((size_t)42 * E * 4);
  u16* memB      = (u16*)alloc((size_t)Smem * E * 2);
  float* qs      = (float*)alloc((size_t)Mdec * E * 4);

  // =================== ENCODER ===================
  gemm(stream, false, zb, encPW, F(ENC_PATCH_B), nullptr, x, nullptr,
       Senc, E, 1024, 1024, E, E, 1.0f, ACT_NONE);
  ln(x, nullptr, nullptr, nullptr, x, Senc);
  addpos_enc_kernel<<<dim3(8192), dim3(256), 0, stream>>>(
      x, F(ENC_SPOS), F(ENC_TPOS), (long long)Senc * E);

  for (int blk = 0; blk < 3; ++blk) {
    int base = ENC_BLK0 + 16 * blk;
    // ---- self attention ----
    ln(x, F(base + EB_LN1G), F(base + EB_LN1B), hb, nullptr, Senc);
    gemm(stream, false, hb, eWq[blk], F(base + EB_BQ), nullptr, nullptr, qb,
         Senc, E, E, E, E, E, 1.0f, ACT_NONE);
    gemm(stream, false, hb, eWk[blk], F(base + EB_BK), nullptr, nullptr, kb,
         Senc, E, E, E, E, E, 1.0f, ACT_NONE);
    gemm(stream, false, hb, eWv[blk], F(base + EB_BV), nullptr, nullptr, vb,
         Senc, E, E, E, E, E, 1.0f, ACT_NONE);
    for (int z = 0; z < 16; ++z) {
      int b = z / Hh, h = z % Hh;
      long long boff = (long long)b * Lenc * E + (long long)h * DH;
      gemm(stream, true, qb + boff, kb + boff, nullptr, nullptr, Sb, nullptr,
           Lenc, Lenc, DH, E, E, Lenc, 0.125f, ACT_NONE);
      softmax_kernel<<<dim3(Lenc, 1), dim3(256), 0, stream>>>(Sb, Pb, Lenc, Lenc, 0, 256);
      gemm(stream, false, Pb, vb + boff, nullptr, nullptr, nullptr, ob + boff,
           Lenc, DH, Lenc, Lenc, E, E, 1.0f, ACT_NONE);
    }
    gemm(stream, false, ob, eWo[blk], F(base + EB_BO), x, x, nullptr,
         Senc, E, E, E, E, E, 1.0f, ACT_NONE);
    // ---- MLP ----
    ln(x, F(base + EB_LN2G), F(base + EB_LN2B), hb, nullptr, Senc);
    gemm(stream, false, hb, eW1[blk], F(base + EB_M_B1), nullptr, nullptr, mlpb,
         Senc, MLPH, E, E, MLPH, MLPH, 1.0f, ACT_GELU);
    gemm(stream, false, mlpb, eW2[blk], F(base + EB_M_B2), x, x, nullptr,
         Senc, E, MLPH, MLPH, E, E, 1.0f, ACT_NONE);
  }

  // final LN + pool + mu + quantize
  ln(x, F(ENC_NORM_G), F(ENC_NORM_B), nullptr, x, Senc);
  meanpool_kernel<<<dim3(16), dim3(256), 0, stream>>>(x, frames, framesB);
  gemm(stream, false, framesB, poolW, F(ENC_POOL_B), nullptr, framesP, nullptr,
       16, E, E, E, E, E, 1.0f, ACT_NONE);
  gather_tr_kernel<<<dim3(14), dim3(256), 0, stream>>>(framesP, trB);
  gemm(stream, false, trB, muW, F(ENC_MU_B), nullptr, mu, nullptr,
       14, 384, E, E, 384, 384, 1.0f, ACT_NONE);
  quantize_kernel<<<dim3(42), dim3(256), 0, stream>>>(mu, F(IN_CODEBOOK), qcB, KCB, CDIM);

  // =================== DECODER MEMORY ===================
  gemm(stream, false, zb, decPW, F(DEC_PATCH_B), nullptr, zp, nullptr,
       1792, E, 1024, 1024, E, E, 1.0f, ACT_NONE,
       2, 1, 2048LL * 1024, 0, 0, 0, 1792LL * E, 0);
  ln(zp, nullptr, nullptr, nullptr, zp, 3584);
  scatter_mem_z_kernel<<<dim3(7168), dim3(256), 0, stream>>>(
      zp, F(DEC_SPOS), F(DEC_TPOS), memB, 3584LL * E);
  gemm(stream, false, qcB, codeW, F(DEC_CODE_B), nullptr, cfull, nullptr,
       42, E, CDIM, CDIM, E, E, 1.0f, ACT_NONE);
  ln(cfull, nullptr, nullptr, nullptr, cfull, 42);
  scatter_mem_c_kernel<<<dim3(84), dim3(256), 0, stream>>>(
      cfull, F(DEC_TPOS), memB, 42LL * E);

  init_query_kernel<<<dim3(1024), dim3(256), 0, stream>>>(
      qs, F(DEC_QEMB), F(DEC_QPOS), (long long)Mdec * E);

  // =================== DECODER BLOCKS ===================
  const long long sQb = 256LL * E;          // per-batch stride in q/k/v [512,512]
  const long long sSs = 256LL * 256;        // per-(b,h) self scores
  const long long sSx = 256LL * LMP;        // per-(b,h) cross scores (padded pitch)
  for (int blk = 0; blk < 3; ++blk) {
    int base = DEC_BLK0 + 26 * blk;
    // ---- self attention ----
    ln(qs, F(base + DB_LN1G), F(base + DB_LN1B), hb, nullptr, Mdec);
    gemm(stream, false, hb, dsWq[blk], F(base + DB_S_BQ), nullptr, nullptr, qb,
         Mdec, E, E, E, E, E, 1.0f, ACT_NONE);
    gemm(stream, false, hb, dsWk[blk], F(base + DB_S_BK), nullptr, nullptr, kb,
         Mdec, E, E, E, E, E, 1.0f, ACT_NONE);
    gemm(stream, false, hb, dsWv[blk], F(base + DB_S_BV), nullptr, nullptr, vb,
         Mdec, E, E, E, E, E, 1.0f, ACT_NONE);
    gemm(stream, true, qb, kb, nullptr, nullptr, Sb, nullptr,
         256, 256, DH, E, E, 256, 0.125f, ACT_NONE,
         16, Hh, sQb, DH, sQb, DH, 8 * sSs, sSs);
    softmax_kernel<<<dim3(256, 16), dim3(256), 0, stream>>>(Sb, Pb, 256, 256, sSs, 0);
    gemm(stream, false, Pb, vb, nullptr, nullptr, nullptr, ob,
         256, DH, 256, 256, E, E, 1.0f, ACT_NONE,
         16, Hh, 8 * sSs, sSs, sQb, DH, sQb, DH);
    gemm(stream, false, ob, dsWo[blk], F(base + DB_S_BO), qs, qs, nullptr,
         Mdec, E, E, E, E, E, 1.0f, ACT_NONE);
    // ---- cross attention ----
    ln(qs, F(base + DB_LN2G), F(base + DB_LN2B), hb, nullptr, Mdec);
    gemm(stream, false, hb, dxWq[blk], F(base + DB_X_BQ), nullptr, nullptr, qb,
         Mdec, E, E, E, E, E, 1.0f, ACT_NONE);
    gemm(stream, false, memB, dxWk[blk], F(base + DB_X_BK), nullptr, nullptr, kb,
         Smem, E, E, E, E, E, 1.0f, ACT_NONE);
    gemm(stream, false, memB, dxWv[blk], F(base + DB_X_BV), nullptr, nullptr, vb,
         Smem, E, E, E, E, E, 1.0f, ACT_NONE);
    gemm(stream, true, qb, kb, nullptr, nullptr, Sb, nullptr,
         256, Lmem, DH, E, E, LMP, 0.125f, ACT_NONE,
         16, Hh, sQb, DH, (long long)Lmem * E, DH, 8 * sSx, sSx);
    softmax_kernel<<<dim3(256, 16), dim3(256), 0, stream>>>(Sb, Pb, Lmem, LMP, sSx, 0);
    gemm(stream, false, Pb, vb, nullptr, nullptr, nullptr, ob,
         256, DH, Lmem, LMP, E, E, 1.0f, ACT_NONE,
         16, Hh, 8 * sSx, sSx, (long long)Lmem * E, DH, sQb, DH);
    gemm(stream, false, ob, dxWo[blk], F(base + DB_X_BO), qs, qs, nullptr,
         Mdec, E, E, E, E, E, 1.0f, ACT_NONE);
    // ---- MLP ----
    ln(qs, F(base + DB_LN3G), F(base + DB_LN3B), hb, nullptr, Mdec);
    gemm(stream, false, hb, dW1[blk], F(base + DB_M_B1), nullptr, nullptr, mlpb,
         Mdec, MLPH, E, E, MLPH, MLPH, 1.0f, ACT_GELU);
    gemm(stream, false, mlpb, dW2[blk], F(base + DB_M_B2), qs, qs, nullptr,
         Mdec, E, MLPH, MLPH, E, E, 1.0f, ACT_NONE);
  }

  // final LN + output projection -> d_out [B,N,D] f32
  ln(qs, F(DEC_NORM_G), F(DEC_NORM_B), hb, nullptr, Mdec);
  gemm(stream, false, hb, outW, F(DEC_OUT_B), nullptr, (float*)d_out, nullptr,
       Mdec, 1024, E, E, 1024, 1024, 1.0f, ACT_NONE);
}
